// LoopBasedConvAttention_11587821765259
// MI455X (gfx1250) — matrix operand order repaired and checked by reading, not compiled, not run on hardware
//
#include <hip/hip_runtime.h>
#include <hip/hip_bf16.h>

// ---------------------------------------------------------------------------
// 3x3 local conv-attention for MI455X (gfx1250, wave32, WMMA bf16->f32).
//  Kernel 1: q/k/v projections  (X[pix,c] @ W[c,d] + b) -> bf16 workspace
//  Kernel 2: 9-neighbor softmax attention (f32 VALU) + ctx @ Wo -> f32 out
// ---------------------------------------------------------------------------

typedef __bf16 bf16;
typedef __attribute__((ext_vector_type(8)))  __bf16 v8bf;
typedef __attribute__((ext_vector_type(16))) __bf16 v16bf;
typedef __attribute__((ext_vector_type(8)))  float  v8f;

#define BATCH 8
#define CCH   128           // C
#define DDIM  128           // D
#define HH    128
#define WW    128
#define HW    (HH * WW)     // 16384
#define LDS_S 136           // padded row stride (elements) -> 272B rows, 16B aligned
#define ROWE  (WW * LDS_S)  // one image-row of LDS rows (17408 elements)

// --- WMMA fragment loaders (ISA 7.12.2 16-bit layouts) ----------------------
// A (16x32, MxK): lane<16 holds row m: K=kb+0..7 (elems0..7), kb+16..23 (8..15)
//                 lane>=16 holds row m: K=kb+8..15,  kb+24..31
__device__ inline v16bf load_frag_a(const bf16* __restrict__ lds,
                                    int m0, int kb, int lane) {
    int half = lane >> 4;
    const bf16* p = lds + (size_t)(m0 + (lane & 15)) * LDS_S + kb + half * 8;
    v8bf lo = *(const v8bf*)p;
    v8bf hi = *(const v8bf*)(p + 16);
    v16bf r;
#pragma unroll
    for (int i = 0; i < 8; ++i) { r[i] = lo[i]; r[i + 8] = hi[i]; }
    return r;
}

// B (32x16, KxN) from K-major LDS (row n holds contiguous K):
//   lane<16 -> column n: K=kb+0..7, kb+16..23 ; lane>=16 -> K=kb+8..15, kb+24..31
__device__ inline v16bf load_frag_b(const bf16* __restrict__ ldsT,
                                    int n0, int kb, int lane) {
    const bf16* p = ldsT + (size_t)(n0 + (lane & 15)) * LDS_S + kb + (lane >> 4) * 8;
    v8bf lo = *(const v8bf*)p;
    v8bf hi = *(const v8bf*)(p + 16);
    v16bf r;
#pragma unroll
    for (int i = 0; i < 8; ++i) { r[i] = lo[i]; r[i + 8] = hi[i]; }
    return r;
}

__device__ inline v8f wmma_bf16(v16bf a, v16bf b, v8f c) {
    return __builtin_amdgcn_wmma_f32_16x16x32_bf16(
        /*neg_a=*/false, a, /*neg_b=*/false, b,
        /*c_mod=*/(short)0, c, /*reuse_a=*/false, /*reuse_b=*/false);
}

// ---------------------------------------------------------------------------
// Kernel 1: q/k/v projection. One block = 128 pixels, 256 threads (8 waves).
// LDS: x-tile [128][LDS_S] bf16 + 3 transposed weights [128][LDS_S] bf16.
// ---------------------------------------------------------------------------
__global__ __launch_bounds__(256)
void qkv_proj_kernel(const float* __restrict__ x,
                     const float* __restrict__ Wq, const float* __restrict__ bq,
                     const float* __restrict__ Wk, const float* __restrict__ bk,
                     const float* __restrict__ Wv, const float* __restrict__ bv,
                     bf16* __restrict__ qo, bf16* __restrict__ ko,
                     bf16* __restrict__ vo) {
    extern __shared__ __align__(16) char smem[];
    bf16* xl = (bf16*)smem;                       // [128][LDS_S]  A operand
    bf16* wt = (bf16*)(smem + 34816);             // 3x [128][LDS_S] (W^T, K-major)

    const int tid = threadIdx.x;
    const int p0  = blockIdx.x * 128;             // global pixel base (never spans b)
    const int b   = p0 >> 14;                     // / HW
    const int hw0 = p0 & (HW - 1);

    // Stage x tile: xl[pixel][c] = x[b][c][hw0+pixel]  (coalesced reads over px)
    for (int i = tid; i < 128 * CCH; i += 256) {
        int c = i >> 7, px = i & 127;
        float v = x[((size_t)(b * CCH + c) << 14) + hw0 + px];
        xl[(size_t)px * LDS_S + c] = (bf16)v;
    }
    // Stage W^T: wt[mat][d][c] = W[c][d]
    const float* Ws[3] = {Wq, Wk, Wv};
#pragma unroll
    for (int m = 0; m < 3; ++m) {
        const float* Wm = Ws[m];
        bf16* dst = wt + m * ROWE;
        for (int i = tid; i < CCH * DDIM; i += 256) {
            int c = i >> 7, d = i & 127;
            dst[(size_t)d * LDS_S + c] = (bf16)Wm[i];
        }
    }
    __syncthreads();

    const int wave = tid >> 5, lane = tid & 31;
    const int m0 = wave * 16;

    // Preload this wave's 4 A fragments (K = 0,32,64,96), reused for all tiles.
    v16bf afr[4];
#pragma unroll
    for (int kb = 0; kb < 4; ++kb) afr[kb] = load_frag_a(xl, m0, kb * 32, lane);

    const float* biases[3] = {bq, bk, bv};
    bf16*        outs[3]   = {qo, ko, vo};
    const int half = lane >> 4;

#pragma unroll
    for (int mat = 0; mat < 3; ++mat) {
        const bf16* wTm = wt + mat * ROWE;
        bf16* op = outs[mat];
#pragma unroll
        for (int nt = 0; nt < 8; ++nt) {
            const int n0 = nt * 16;
            v8f acc = {};
#pragma unroll
            for (int kb = 0; kb < 4; ++kb) {
                v16bf bfr = load_frag_b(wTm, n0, kb * 32, lane);
                acc = wmma_bf16(afr[kb], bfr, acc);
            }
            const int d = n0 + (lane & 15);            // N index (fixed per lane)
            const float bias = biases[mat][d];
#pragma unroll
            for (int r = 0; r < 8; ++r) {
                int pix = p0 + m0 + r + 8 * half;      // M index
                op[(size_t)pix * DDIM + d] = (bf16)(acc[r] + bias);
            }
        }
    }
}

// ---------------------------------------------------------------------------
// Kernel 2: per-(b,h) row. Stage q row + 3 k/v rows in LDS (bf16), 9-neighbor
// masked softmax in f32, ctx written back into q buffer, then ctx @ Wo via
// WMMA with M=cout / N=w so f32 stores are w-contiguous across lanes.
// ---------------------------------------------------------------------------
__global__ __launch_bounds__(256)
void attn_out_kernel(const bf16* __restrict__ qw, const bf16* __restrict__ kw,
                     const bf16* __restrict__ vw, const float* __restrict__ Wo,
                     const float* __restrict__ bo, float* __restrict__ out) {
    extern __shared__ __align__(16) char smem[];
    bf16* ql  = (bf16*)smem;                 // [128][LDS_S]  q row, then ctx
    bf16* kl  = (bf16*)(smem + 34816);       // [3][128][LDS_S]
    bf16* vl  = (bf16*)(smem + 139264);      // [3][128][LDS_S]
    bf16* wot = (bf16*)(smem + 243712);      // Wo^T [cout][d] (K-major A operand)

    const int tid = threadIdx.x;
    const int b = blockIdx.x >> 7;
    const int h = blockIdx.x & 127;

    // q row (16B vector copies), LDS rows padded to LDS_S
    {
        const uint4* src = (const uint4*)(qw + ((size_t)blockIdx.x << 14));
        for (int i = tid; i < 2048; i += 256) {
            int w = i >> 4, c = i & 15;
            *(uint4*)(ql + (size_t)w * LDS_S + c * 8) = src[i];
        }
    }
    // k/v rows h-1..h+1 (skip out-of-range rows; never read by masked lanes)
#pragma unroll
    for (int r = 0; r < 3; ++r) {
        int hh = h - 1 + r;
        if (hh < 0 || hh >= HH) continue;
        const size_t base = ((size_t)(b * HH + hh) << 14);
        const uint4* ks = (const uint4*)(kw + base);
        const uint4* vs = (const uint4*)(vw + base);
        bf16* kd = kl + r * ROWE;
        bf16* vd = vl + r * ROWE;
        for (int i = tid; i < 2048; i += 256) {
            int w = i >> 4, c = i & 15;
            *(uint4*)(kd + (size_t)w * LDS_S + c * 8) = ks[i];
            *(uint4*)(vd + (size_t)w * LDS_S + c * 8) = vs[i];
        }
    }
    // Wo^T: wot[cout][d] = Wo[d][cout]
    for (int i = tid; i < DDIM * CCH; i += 256) {
        int d = i >> 7, c = i & 127;
        wot[(size_t)c * LDS_S + d] = (bf16)Wo[i];
    }
    __syncthreads();

    // ---- Phase A: 9-neighbor masked softmax attention (one pixel / thread)
    if (tid < WW) {
        const int w = tid;
        const float scale = 0.0883883476483184f;   // 1/sqrt(128)
        float s[9];
        bool  ok[9];
#pragma unroll
        for (int kk = 0; kk < 9; ++kk) {
            const int di = kk / 3 - 1, dj = kk % 3 - 1;
            const int hh = h + di, ww = w + dj;
            const bool v = (hh >= 0) && (hh < HH) && (ww >= 0) && (ww < WW);
            ok[kk] = v;
            float acc = 0.f;
            if (v) {
                const bf16* qp = ql + (size_t)w * LDS_S;
                const bf16* kp = kl + (size_t)(di + 1) * ROWE + (size_t)ww * LDS_S;
                for (int d8 = 0; d8 < 16; ++d8) {
                    v8bf qv = *(const v8bf*)(qp + d8 * 8);
                    v8bf kv = *(const v8bf*)(kp + d8 * 8);
#pragma unroll
                    for (int e = 0; e < 8; ++e)
                        acc = __builtin_fmaf((float)qv[e], (float)kv[e], acc);
                }
            }
            s[kk] = acc * scale;
        }
        float mx = -1e30f;
#pragma unroll
        for (int kk = 0; kk < 9; ++kk) if (ok[kk] && s[kk] > mx) mx = s[kk];
        float wgt[9], sum = 0.f;
#pragma unroll
        for (int kk = 0; kk < 9; ++kk) {
            float e = ok[kk] ? __expf(s[kk] - mx) : 0.f;
            wgt[kk] = e; sum += e;
        }
        const float inv = 1.f / sum;
        // ctx (overwrites this thread's own q row only)
        for (int d8 = 0; d8 < 16; ++d8) {
            float c8[8];
#pragma unroll
            for (int e = 0; e < 8; ++e) c8[e] = 0.f;
#pragma unroll
            for (int kk = 0; kk < 9; ++kk) {
                if (!ok[kk]) continue;
                const int di = kk / 3 - 1, dj = kk % 3 - 1;
                const bf16* vp = vl + (size_t)(di + 1) * ROWE +
                                 (size_t)(w + dj) * LDS_S + d8 * 8;
                v8bf vv = *(const v8bf*)vp;
#pragma unroll
                for (int e = 0; e < 8; ++e)
                    c8[e] = __builtin_fmaf(wgt[kk], (float)vv[e], c8[e]);
            }
            bf16* cp = ql + (size_t)w * LDS_S + d8 * 8;
#pragma unroll
            for (int e = 0; e < 8; ++e) cp[e] = (bf16)(c8[e] * inv);
        }
    }
    __syncthreads();

    // ---- Phase B: out[b,cout,h,w] = ctx[w,:] @ Wo[:,cout] + bo[cout]
    // A = Wo^T (M=cout, K=d), B = ctx (K=d, N=w). 64 tiles over 8 waves.
    const int wave = tid >> 5, lane = tid & 31;
    const int half = lane >> 4;
#pragma unroll
    for (int t = 0; t < 8; ++t) {
        const int tile = wave * 8 + t;
        const int m0 = (tile >> 3) * 16;   // cout base
        const int n0 = (tile & 7) * 16;    // w base
        v8f acc = {};
#pragma unroll
        for (int kb = 0; kb < 4; ++kb) {
            v16bf a  = load_frag_a(wot, m0, kb * 32, lane);
            v16bf bb = load_frag_b(ql,  n0, kb * 32, lane);
            acc = wmma_bf16(a, bb, acc);
        }
        const int w = n0 + (lane & 15);    // N per lane -> contiguous stores
#pragma unroll
        for (int r = 0; r < 8; ++r) {
            const int cout = m0 + r + 8 * half;
            out[((size_t)(b * CCH + cout) << 14) + (h << 7) + w] =
                acc[r] + bo[cout];
        }
    }
}

// ---------------------------------------------------------------------------
extern "C" void kernel_launch(void* const* d_in, const int* in_sizes, int n_in,
                              void* d_out, int out_size, void* d_ws, size_t ws_size,
                              hipStream_t stream) {
    const float* x  = (const float*)d_in[0];
    const float* Wq = (const float*)d_in[1];
    const float* bq = (const float*)d_in[2];
    const float* Wk = (const float*)d_in[3];
    const float* bk = (const float*)d_in[4];
    const float* Wv = (const float*)d_in[5];
    const float* bv = (const float*)d_in[6];
    const float* Wo = (const float*)d_in[7];
    const float* bo = (const float*)d_in[8];
    float* out = (float*)d_out;

    const size_t npix = (size_t)BATCH * HW;            // 131072
    const size_t qkv_bytes = npix * DDIM * sizeof(bf16); // 32 MiB each
    bf16* qws = (bf16*)d_ws;
    bf16* kws = (bf16*)((char*)d_ws + qkv_bytes);
    bf16* vws = (bf16*)((char*)d_ws + 2 * qkv_bytes);

    const int blocks1 = (int)(npix / 128);             // 1024
    const size_t lds1 = 139264;                        // x tile + 3 W^T (bf16)
    qkv_proj_kernel<<<blocks1, 256, lds1, stream>>>(
        x, Wq, bq, Wk, bk, Wv, bv, qws, kws, vws);

    const int blocks2 = BATCH * HH;                    // 1024
    const size_t lds2 = 278528;                        // q/ctx + 3k + 3v + Wo^T
    attn_out_kernel<<<blocks2, 256, lds2, stream>>>(
        qws, kws, vws, Wo, bo, out);
}
